// PreT_Attention_67697274519701
// MI455X (gfx1250) — matrix operand order repaired and checked by reading, not compiled, not run on hardware
//
#include <hip/hip_runtime.h>
#include <hip/hip_bf16.h>

// ---------------------------------------------------------------------------
// PreT attention for MI455X (gfx1250, wave32, WMMA bf16 16x16x32, f32 accum)
// GEMMs use 32x64 output tiles per wave (8 accumulators) for A/B frag reuse.
// ---------------------------------------------------------------------------

typedef __bf16 bf16_t;
typedef __attribute__((ext_vector_type(16))) __bf16 v16bf;
typedef __attribute__((ext_vector_type(8)))  __bf16 v8bf;
typedef __attribute__((ext_vector_type(8)))  float  v8f;

static constexpr int B_   = 8;
static constexpr int N_   = 1024;
static constexpr int C_   = 1024;
static constexpr int P_   = 64;
static constexpr int H_   = 16;
static constexpr int HD_  = 64;
static constexpr int LTOT = P_ + N_;   // 1088

// ------------------------------ WMMA helpers -------------------------------

__device__ __forceinline__ v8f zero8() {
  v8f z;
#pragma unroll
  for (int i = 0; i < 8; ++i) z[i] = 0.0f;
  return z;
}

__device__ __forceinline__ v16bf combine16(v8bf lo, v8bf hi) {
  v16bf r;
#pragma unroll
  for (int i = 0; i < 8; ++i) { r[i] = lo[i]; r[i + 8] = hi[i]; }
  return r;
}

// A operand, 16x32 bf16, row-major source [M x K] with leading dim lda.
// Lane (half = lane>>4, m = lane&15) holds row m0+m:
//   elems 0..7  = K[k0 + 8*half .. +7], elems 8..15 = K[k0 + 16 + 8*half ..]
__device__ __forceinline__ v16bf load_a_frag(const bf16_t* __restrict__ A,
                                             int lda, int m0, int k0) {
  const int lane = threadIdx.x & 31;
  const int half = lane >> 4;
  const int m    = lane & 15;
  const bf16_t* p = A + (size_t)(m0 + m) * lda + k0 + 8 * half;
  v8bf lo = *(const v8bf*)p;
  v8bf hi = *(const v8bf*)(p + 16);
  return combine16(lo, hi);
}

// B operand, 32x16 bf16, source stored transposed (row n = output column n,
// contiguous over reduction dim) with leading dim ldb.
// Lane (half, n): elems 0..7 = K[k0 + 8*half ..], elems 8..15 = K[k0 + 16 + 8*half ..].
__device__ __forceinline__ v16bf load_b_frag(const bf16_t* __restrict__ Bt,
                                             int ldb, int n0, int k0) {
  const int lane = threadIdx.x & 31;
  const int half = lane >> 4;
  const int n    = lane & 15;
  const bf16_t* p = Bt + (size_t)(n0 + n) * ldb + k0 + 8 * half;
  v8bf lo = *(const v8bf*)p;
  v8bf hi = *(const v8bf*)(p + 16);
  return combine16(lo, hi);
}

__device__ __forceinline__ v8f wmma_bf16(v16bf a, v16bf b, v8f c) {
  return __builtin_amdgcn_wmma_f32_16x16x32_bf16(false, a, false, b,
                                                 (short)0, c, false, false);
}

// ------------------------------ small kernels ------------------------------

__global__ void cvt_f32_bf16(const float* __restrict__ in,
                             bf16_t* __restrict__ out, int n) {
  int i = blockIdx.x * blockDim.x + threadIdx.x;
  if (i < n) out[i] = (bf16_t)in[i];
}

// prompt [B,2,P,H,hd] f32 -> k_bf prefix rows (*alpha), vt_bf prefix cols (*alpha)
__global__ void prefix_fill(const float* __restrict__ prompt,
                            const float* __restrict__ alpha,
                            bf16_t* __restrict__ kbf,
                            bf16_t* __restrict__ vtbf) {
  int i = blockIdx.x * blockDim.x + threadIdx.x;   // over B*H*P*HD = 524288
  if (i >= B_ * H_ * P_ * HD_) return;
  int d = i & 63;
  int p = (i >> 6) & 63;
  int h = (i >> 12) & 15;
  int b = i >> 16;
  float a = alpha[0];
  float kv = prompt[((((size_t)b * 2 + 0) * P_ + p) * H_ + h) * HD_ + d];
  float vv = prompt[((((size_t)b * 2 + 1) * P_ + p) * H_ + h) * HD_ + d];
  kbf [((size_t)(b * H_ + h) * LTOT + p) * HD_ + d]   = (bf16_t)(kv * a);
  vtbf[((size_t)(b * H_ + h) * HD_ + d) * LTOT + p]   = (bf16_t)(vv * a);
}

// ------------------------------ QKV GEMM -----------------------------------
// qkv = x @ Wqkv^T ; 32x64 tile per wave; 64-col tile == one head's d-range.
// Epilogue scatters into q (*scale), k (*ha*a), v^T (*ha*a).
__global__ void __launch_bounds__(256)
qkv_gemm(const bf16_t* __restrict__ xbf, const bf16_t* __restrict__ wbf,
         const float* __restrict__ alpha, const float* __restrict__ half_alpha,
         bf16_t* __restrict__ qbf, bf16_t* __restrict__ kbf,
         bf16_t* __restrict__ vtbf) {
  const int COLT = (3 * C_) / 64;                       // 48
  int wave = blockIdx.x * (blockDim.x >> 5) + (threadIdx.x >> 5);
  int rt = wave / COLT, ct = wave % COLT;
  int r0 = rt * 32, j0 = ct * 64;

  v8f acc[2][4];
#pragma unroll
  for (int rr = 0; rr < 2; ++rr)
#pragma unroll
    for (int t = 0; t < 4; ++t) acc[rr][t] = zero8();

#pragma unroll 2
  for (int k0 = 0; k0 < C_; k0 += 32) {
    __builtin_prefetch(xbf + (size_t)(r0 + (threadIdx.x & 31)) * C_ + k0 + 64, 0, 3);
    __builtin_prefetch(wbf + (size_t)(j0 + (threadIdx.x & 31)) * C_ + k0 + 64, 0, 3);
    v16bf a0 = load_a_frag(xbf, C_, r0,      k0);
    v16bf a1 = load_a_frag(xbf, C_, r0 + 16, k0);
#pragma unroll
    for (int t = 0; t < 4; ++t) {
      v16bf b = load_b_frag(wbf, C_, j0 + 16 * t, k0);
      acc[0][t] = wmma_bf16(a0, b, acc[0][t]);
      acc[1][t] = wmma_bf16(a1, b, acc[1][t]);
    }
  }

  const int lane = threadIdx.x & 31;
  const int half = lane >> 4;
  const int cn   = lane & 15;
  int sec  = j0 / C_;                 // 64 | 1024, tile within one section
  int h    = (j0 % C_) / HD_;         // tile covers exactly head h, d = 0..63
  int bidx = r0 / N_;
  int n0   = r0 % N_;
  float sc = (sec == 0) ? 0.125f /* hd^-0.5 */ : half_alpha[0] * alpha[0];
#pragma unroll
  for (int rr = 0; rr < 2; ++rr) {
#pragma unroll
    for (int t = 0; t < 4; ++t) {
      int d = t * 16 + cn;
#pragma unroll
      for (int r = 0; r < 8; ++r) {
        int n = n0 + rr * 16 + half * 8 + r;
        bf16_t bv = (bf16_t)(acc[rr][t][r] * sc);
        if (sec == 0)
          qbf[((size_t)(bidx * H_ + h) * N_ + n) * HD_ + d] = bv;
        else if (sec == 1)
          kbf[((size_t)(bidx * H_ + h) * LTOT + P_ + n) * HD_ + d] = bv;
        else
          vtbf[((size_t)(bidx * H_ + h) * HD_ + d) * LTOT + P_ + n] = bv;
      }
    }
  }
}

// ---------------------------- flash attention ------------------------------
// One wave owns a 16-row Q tile; streams 32 keys/iter; online softmax;
// P tile round-trips per-wave LDS to become an A-operand fragment.
__global__ void __launch_bounds__(256)
attn_kernel(const bf16_t* __restrict__ qbf, const bf16_t* __restrict__ kbf,
            const bf16_t* __restrict__ vtbf, bf16_t* __restrict__ attn_bf) {
  __shared__ bf16_t ptile[8][16 * 32];
  const int wslot = threadIdx.x >> 5;
  const int wave  = blockIdx.x * 8 + wslot;
  const int QT    = N_ / 16;                 // 64 q-tiles per (b,h)
  const int qt    = wave % QT;
  const int bh    = wave / QT;               // 0..127
  const int lane  = threadIdx.x & 31;
  const int half  = lane >> 4;
  const int ln    = lane & 15;
  const int rbase = half * 8;

  const bf16_t* Q  = qbf  + (size_t)bh * N_   * HD_;
  const bf16_t* K  = kbf  + (size_t)bh * LTOT * HD_;
  const bf16_t* Vt = vtbf + (size_t)bh * HD_  * LTOT;
  const int q0 = qt * 16;

  v16bf qf0 = load_a_frag(Q, HD_, q0, 0);
  v16bf qf1 = load_a_frag(Q, HD_, q0, 32);

  v8f o0 = zero8(), o1 = zero8(), o2 = zero8(), o3 = zero8();
  float mrow[8], lrow[8];
#pragma unroll
  for (int r = 0; r < 8; ++r) { mrow[r] = -1e30f; lrow[r] = 0.0f; }

  bf16_t* myP = ptile[wslot];

  for (int j0 = 0; j0 < LTOT; j0 += 32) {
    // prefetch next chunk of K rows and V^T columns
    __builtin_prefetch(K  + (size_t)(j0 + 32 + lane) * HD_, 0, 3);
    __builtin_prefetch(Vt + (size_t)lane * LTOT + j0 + 32, 0, 3);

    // ---- S = (q*scale) . K^T for 16q x 32k ----
    v8f s0 = zero8(), s1 = zero8();
    s0 = wmma_bf16(qf0, load_b_frag(K, HD_, j0,      0),  s0);
    s0 = wmma_bf16(qf1, load_b_frag(K, HD_, j0,      32), s0);
    s1 = wmma_bf16(qf0, load_b_frag(K, HD_, j0 + 16, 0),  s1);
    s1 = wmma_bf16(qf1, load_b_frag(K, HD_, j0 + 16, 32), s1);

    // ---- online softmax (16-lane butterfly per half-wave) ----
    float mn[8], rs[8];
#pragma unroll
    for (int r = 0; r < 8; ++r) {
      float cm = fmaxf(s0[r], s1[r]);
      cm = fmaxf(cm, __shfl_xor(cm, 1));
      cm = fmaxf(cm, __shfl_xor(cm, 2));
      cm = fmaxf(cm, __shfl_xor(cm, 4));
      cm = fmaxf(cm, __shfl_xor(cm, 8));
      mn[r] = fmaxf(mrow[r], cm);
      float p0 = __expf(s0[r] - mn[r]);
      float p1 = __expf(s1[r] - mn[r]);
      s0[r] = p0; s1[r] = p1;
      float sum = p0 + p1;
      sum += __shfl_xor(sum, 1);
      sum += __shfl_xor(sum, 2);
      sum += __shfl_xor(sum, 4);
      sum += __shfl_xor(sum, 8);
      rs[r] = sum;
    }
#pragma unroll
    for (int r = 0; r < 8; ++r) {
      float corr = __expf(mrow[r] - mn[r]);
      lrow[r] = lrow[r] * corr + rs[r];
      mrow[r] = mn[r];
      o0[r] *= corr; o1[r] *= corr; o2[r] *= corr; o3[r] *= corr;
      int row = rbase + r;
      myP[row * 32 + ln]      = (bf16_t)s0[r];
      myP[row * 32 + 16 + ln] = (bf16_t)s1[r];
    }
    asm volatile("s_wait_dscnt 0x0" ::: "memory");  // intra-wave LDS RAW

    // re-read P in A-fragment layout
    v8bf plo = *(const v8bf*)(myP + ln * 32 + 8 * half);
    v8bf phi = *(const v8bf*)(myP + ln * 32 + 16 + 8 * half);
    v16bf pf = combine16(plo, phi);

    // ---- O += P . V  (B frags from transposed V: contiguous loads) ----
    o0 = wmma_bf16(pf, load_b_frag(Vt, LTOT, 0,  j0), o0);
    o1 = wmma_bf16(pf, load_b_frag(Vt, LTOT, 16, j0), o1);
    o2 = wmma_bf16(pf, load_b_frag(Vt, LTOT, 32, j0), o2);
    o3 = wmma_bf16(pf, load_b_frag(Vt, LTOT, 48, j0), o3);
  }

  // epilogue: O/l  -> attn_bf [B, N, C] (head-interleaved for out-proj)
  const int b = bh / H_, h = bh % H_;
#pragma unroll
  for (int r = 0; r < 8; ++r) {
    int n = q0 + rbase + r;
    float inv = 1.0f / lrow[r];
    size_t base = (size_t)(b * N_ + n) * C_ + h * HD_;
    attn_bf[base +  0 + ln] = (bf16_t)(o0[r] * inv);
    attn_bf[base + 16 + ln] = (bf16_t)(o1[r] * inv);
    attn_bf[base + 32 + ln] = (bf16_t)(o2[r] * inv);
    attn_bf[base + 48 + ln] = (bf16_t)(o3[r] * inv);
  }
}

// ------------------------------ out-proj GEMM ------------------------------
// 32x64 tile per wave, bias add, f32 output.
__global__ void __launch_bounds__(256)
proj_gemm(const bf16_t* __restrict__ abf, const bf16_t* __restrict__ wbf,
          const float* __restrict__ bias, float* __restrict__ out) {
  const int COLT = C_ / 64;                           // 16
  int wave = blockIdx.x * (blockDim.x >> 5) + (threadIdx.x >> 5);
  int rt = wave / COLT, ct = wave % COLT;
  int r0 = rt * 32, j0 = ct * 64;

  v8f acc[2][4];
#pragma unroll
  for (int rr = 0; rr < 2; ++rr)
#pragma unroll
    for (int t = 0; t < 4; ++t) acc[rr][t] = zero8();

#pragma unroll 2
  for (int k0 = 0; k0 < C_; k0 += 32) {
    __builtin_prefetch(abf + (size_t)(r0 + (threadIdx.x & 31)) * C_ + k0 + 64, 0, 3);
    __builtin_prefetch(wbf + (size_t)(j0 + (threadIdx.x & 31)) * C_ + k0 + 64, 0, 3);
    v16bf a0 = load_a_frag(abf, C_, r0,      k0);
    v16bf a1 = load_a_frag(abf, C_, r0 + 16, k0);
#pragma unroll
    for (int t = 0; t < 4; ++t) {
      v16bf b = load_b_frag(wbf, C_, j0 + 16 * t, k0);
      acc[0][t] = wmma_bf16(a0, b, acc[0][t]);
      acc[1][t] = wmma_bf16(a1, b, acc[1][t]);
    }
  }
  const int lane = threadIdx.x & 31;
  const int half = lane >> 4;
  const int cn   = lane & 15;
#pragma unroll
  for (int t = 0; t < 4; ++t) {
    float bv = bias[j0 + 16 * t + cn];
#pragma unroll
    for (int rr = 0; rr < 2; ++rr)
#pragma unroll
      for (int r = 0; r < 8; ++r)
        out[(size_t)(r0 + rr * 16 + half * 8 + r) * C_ + j0 + 16 * t + cn] =
            acc[rr][t][r] + bv;
  }
}

// ------------------------------- launcher ----------------------------------

extern "C" void kernel_launch(void* const* d_in, const int* in_sizes, int n_in,
                              void* d_out, int out_size, void* d_ws,
                              size_t ws_size, hipStream_t stream) {
  (void)in_sizes; (void)n_in; (void)out_size; (void)ws_size;
  const float* x          = (const float*)d_in[0];
  const float* prompt     = (const float*)d_in[1];
  const float* alpha      = (const float*)d_in[2];
  const float* half_alpha = (const float*)d_in[3];
  const float* Wqkv       = (const float*)d_in[4];
  const float* Wproj      = (const float*)d_in[5];
  const float* bproj      = (const float*)d_in[6];
  float* out = (float*)d_out;

  char* ws = (char*)d_ws;
  size_t off = 0;
  auto alloc = [&](size_t bytes) -> void* {
    void* p = ws + off;
    off += (bytes + 255) & ~(size_t)255;
    return p;
  };
  bf16_t* xbf   = (bf16_t*)alloc((size_t)B_ * N_ * C_ * 2);     // 16 MB
  bf16_t* wqkvb = (bf16_t*)alloc((size_t)3 * C_ * C_ * 2);      //  6 MB
  bf16_t* wprob = (bf16_t*)alloc((size_t)C_ * C_ * 2);          //  2 MB
  bf16_t* qbf   = (bf16_t*)alloc((size_t)B_ * H_ * N_ * HD_ * 2);
  bf16_t* kbf   = (bf16_t*)alloc((size_t)B_ * H_ * LTOT * HD_ * 2);
  bf16_t* vtbf  = (bf16_t*)alloc((size_t)B_ * H_ * HD_ * LTOT * 2);
  bf16_t* attnb = xbf;   // x_bf dead after QKV GEMM -> reuse for attn output

  // 1) f32 -> bf16 conversions
  {
    int n = B_ * N_ * C_;                                   // 8388608
    cvt_f32_bf16<<<(n + 255) / 256, 256, 0, stream>>>(x, xbf, n);
    n = 3 * C_ * C_;                                        // 3145728
    cvt_f32_bf16<<<(n + 255) / 256, 256, 0, stream>>>(Wqkv, wqkvb, n);
    n = C_ * C_;                                            // 1048576
    cvt_f32_bf16<<<(n + 255) / 256, 256, 0, stream>>>(Wproj, wprob, n);
  }
  // 2) prefix K/V (scaled by alpha)
  {
    int n = B_ * H_ * P_ * HD_;                             // 524288
    prefix_fill<<<(n + 255) / 256, 256, 0, stream>>>(prompt, alpha, kbf, vtbf);
  }
  // 3) QKV GEMM: (8192/32)*(3072/64) = 12288 wave-tiles, 8 waves/block
  qkv_gemm<<<12288 / 8, 256, 0, stream>>>(xbf, wqkvb, alpha, half_alpha,
                                          qbf, kbf, vtbf);
  // 4) flash attention: 8*16*(1024/16) = 8192 waves
  attn_kernel<<<8192 / 8, 256, 0, stream>>>(qbf, kbf, vtbf, attnb);
  // 5) out-proj: (8192/32)*(1024/64) = 4096 wave-tiles
  proj_gemm<<<4096 / 8, 256, 0, stream>>>(attnb, wprob, bproj, out);
}
